// Model_48026324304474
// MI455X (gfx1250) — matrix operand order repaired and checked by reading, not compiled, not run on hardware
//
#include <hip/hip_runtime.h>
#include <hip/hip_bf16.h>
#include <math.h>

// ---------------- problem constants ----------------
constexpr int B = 16, T = 750, D = 2048, C = 20;
constexpr int TP = T + 2;                 // zero-padded rows per batch (conv pad=1)
constexpr int KDIM = 3 * D;               // 6144 (im2col K)
constexpr int K_ACT = 83;                 // T // 9
constexpr int K_BKG = 187;                // T // 4
constexpr long XP_ROWS = (long)B * TP + 2; // + 2 slack rows (zero) for M-tile overshoot

typedef __attribute__((ext_vector_type(16))) __bf16 v16bf;
typedef __attribute__((ext_vector_type(8)))  float  v8f;

union Frag { v16bf v; uint4 u[2]; };

static __device__ __forceinline__ unsigned short f2bf(float f) {
  unsigned int u = __float_as_uint(f);
  unsigned int r = (u + 0x7FFFu + ((u >> 16) & 1u)) >> 16;   // round-to-nearest-even
  return (unsigned short)r;
}

// ---------------- prep kernels ----------------
// Xp[b][tt][d] = bf16(x[b][tt-1][d]) with tt==0 / tt==T+1 zero pads, plus 2 zero slack rows.
__global__ void prep_x(const float* __restrict__ x, unsigned short* __restrict__ xp) {
  long i = (long)blockIdx.x * 256 + threadIdx.x;
  long total = XP_ROWS * (long)D;
  if (i >= total) return;
  long r = i >> 11; int d = (int)(i & 2047);
  unsigned short v = 0;
  if (r < (long)B * TP) {
    int b = (int)(r / TP), tt = (int)(r % TP);
    if (tt >= 1 && tt <= T) v = f2bf(x[((long)b * T + (tt - 1)) * D + d]);
  }
  xp[i] = v;
}

// WpT[co][dk*2048+ci] = bf16(conv_w[co][ci][dk])  -> B-fragment rows are contiguous in K
__global__ void prep_w(const float* __restrict__ w, unsigned short* __restrict__ wpt) {
  int i = blockIdx.x * 256 + threadIdx.x;
  if (i >= D * KDIM) return;
  int co = i / KDIM; int r = i - co * KDIM;
  int dk = r >> 11; int ci = r & 2047;
  wpt[i] = f2bf(w[((long)co * D + ci) * 3 + dk]);
}

// fused BN affine: y = relu(acc * scale + offs)
__global__ void prep_bn(const float* __restrict__ g, const float* __restrict__ be,
                        const float* __restrict__ mn, const float* __restrict__ vr,
                        const float* __restrict__ cb,
                        float* __restrict__ scale, float* __restrict__ offs) {
  int i = blockIdx.x * 256 + threadIdx.x;
  if (i >= D) return;
  float s = g[i] * rsqrtf(vr[i] + 1e-5f);
  scale[i] = s;
  offs[i] = (cb[i] - mn[i]) * s + be[i];
}

// ---------------- main conv-as-GEMM with BF16 WMMA ----------------
// Block: 512 threads = 16 waves arranged 4(M) x 4(N); per-wave tile 32M x 64N.
// Block tile 128M x 256N.  K = 6144 in chunks of 32 (one v_wmma_f32_16x16x32_bf16 each).
// L2 read volume: (12000/128)*|W| + (2048/256)*|X| ~= 3.5 GB (vs 5.9 GB at BM=64).
__global__ void __launch_bounds__(512)
conv_gemm(const unsigned short* __restrict__ xp, const unsigned short* __restrict__ wpt,
          const float* __restrict__ scale, const float* __restrict__ offs,
          float* __restrict__ feat) {
  const int lane  = threadIdx.x & 31;
  const int wave  = threadIdx.x >> 5;   // 0..15
  const int laneH = lane >> 4;          // 0 = lanes 0-15, 1 = lanes 16-31
  const int laneM = lane & 15;
  const int mOff  = (wave & 3) * 32;    // 4 M-sub waves
  const int nOff  = (wave >> 2) * 64;   // 4 N-sub waves
  const int b     = blockIdx.x / 6;
  const int tBase = (blockIdx.x % 6) * 128;
  const int nBase = blockIdx.y * 256;

  // A row pointers: im2col row for (b,t) = 6144 contiguous bf16 at padded row t.
  const unsigned short* aPtr[2];
#pragma unroll
  for (int mt = 0; mt < 2; ++mt) {
    long prow = (long)b * TP + tBase + mOff + mt * 16 + laneM;
    long pmax = (long)B * TP - 1;                 // +2 slack rows cover the 3-row span
    if (prow > pmax) prow = pmax;                 // overshoot rows: computed, never stored
    aPtr[mt] = xp + prow * (long)D;
  }
  const unsigned short* bPtr[4];
#pragma unroll
  for (int nt = 0; nt < 4; ++nt) {
    int n = nBase + nOff + nt * 16 + laneM;
    bPtr[nt] = wpt + (long)n * KDIM;
  }

  v8f acc[2][4];
#pragma unroll
  for (int mt = 0; mt < 2; ++mt)
#pragma unroll
    for (int nt = 0; nt < 4; ++nt) acc[mt][nt] = (v8f){0.f,0.f,0.f,0.f,0.f,0.f,0.f,0.f};

  for (int kc = 0; kc < KDIM; kc += 32) {
    Frag a[2], bb[4];
    // A 16x32 bf16 layout: lo lanes K{0-7,16-23}, hi lanes K{8-15,24-31}
    const int kA = kc + laneH * 8;
#pragma unroll
    for (int mt = 0; mt < 2; ++mt) {
      const unsigned short* p = aPtr[mt] + kA;
      a[mt].u[0] = *(const uint4*)(p);
      a[mt].u[1] = *(const uint4*)(p + 16);
    }
    // B 32x16 bf16 layout: lo lanes K{0-7,16-23}, hi lanes K{8-15,24-31}
    const int kB = kc + laneH * 8;
#pragma unroll
    for (int nt = 0; nt < 4; ++nt) {
      const unsigned short* p = bPtr[nt] + kB;
      bb[nt].u[0] = *(const uint4*)(p);
      bb[nt].u[1] = *(const uint4*)(p + 16);
    }
#pragma unroll
    for (int mt = 0; mt < 2; ++mt)
#pragma unroll
      for (int nt = 0; nt < 4; ++nt)
        acc[mt][nt] = __builtin_amdgcn_wmma_f32_16x16x32_bf16(
            false, a[mt].v, false, bb[nt].v, (short)0, acc[mt][nt], false, false);
  }

  // Epilogue: BN affine + ReLU, write features [B,T,2048].
  // C/D layout: VGPR r -> M = r + 8*laneH, N = laneM.
#pragma unroll
  for (int nt = 0; nt < 4; ++nt) {
    int n = nBase + nOff + nt * 16 + laneM;
    float s = scale[n], o = offs[n];
#pragma unroll
    for (int mt = 0; mt < 2; ++mt) {
      int t0 = tBase + mOff + mt * 16 + laneH * 8;
#pragma unroll
      for (int r = 0; r < 8; ++r) {
        int t = t0 + r;
        if (t < T) {
          float v = fmaf(acc[mt][nt][r], s, o);
          feat[((long)b * T + t) * D + n] = v > 0.f ? v : 0.f;
        }
      }
    }
  }
}

// ---------------- cas + feat_mag + cas_softmax (one wave per (b,t) row) ----------------
__global__ void __launch_bounds__(256)
cas_mag(const float* __restrict__ feat, const float* __restrict__ clsw,
        float* __restrict__ cas, float* __restrict__ fmag, float* __restrict__ cas_sm) {
  int rt = blockIdx.x * 8 + (threadIdx.x >> 5);
  if (rt >= B * T) return;
  int lane = threadIdx.x & 31;
  const float* row = feat + (long)rt * D;
  float mag2 = 0.f;
  float a[C];
#pragma unroll
  for (int c = 0; c < C; ++c) a[c] = 0.f;
  for (int i = 0; i < D / 32; ++i) {
    int k = i * 32 + lane;
    float f = row[k];
    mag2 = fmaf(f, f, mag2);
#pragma unroll
    for (int c = 0; c < C; ++c) a[c] = fmaf(f, clsw[c * D + k], a[c]);
  }
#pragma unroll
  for (int off = 16; off > 0; off >>= 1) {
    mag2 += __shfl_xor(mag2, off, 32);
#pragma unroll
    for (int c = 0; c < C; ++c) a[c] += __shfl_xor(a[c], off, 32);
  }
  if (lane == 0) {
    fmag[rt] = sqrtf(mag2);
    float mx = a[0];
#pragma unroll
    for (int c = 1; c < C; ++c) mx = fmaxf(mx, a[c]);
    float ssum = 0.f; float e[C];
#pragma unroll
    for (int c = 0; c < C; ++c) { e[c] = __expf(a[c] - mx); ssum += e[c]; }
    float inv = 1.f / ssum;
#pragma unroll
    for (int c = 0; c < C; ++c) {
      cas[(long)rt * C + c] = a[c];
      cas_sm[(long)rt * C + c] = e[c] * inv;
    }
  }
}

// ---------------- exact stable top-k index selection via rank counting ----------------
__global__ void __launch_bounds__(256)
topk_idx(const float* __restrict__ fmag, int* __restrict__ idx_act, int* __restrict__ idx_bkg) {
  __shared__ float m[T];
  int b = blockIdx.x;
  for (int t = threadIdx.x; t < T; t += 256) m[t] = fmag[b * T + t];
  __syncthreads();
  for (int t = threadIdx.x; t < T; t += 256) {
    float v = m[t];
    int rd = 0, ra = 0;
    for (int tp = 0; tp < T; ++tp) {
      float vp = m[tp];
      bool tie = (vp == v) && (tp < t);
      rd += ((vp > v) || tie) ? 1 : 0;   // position in stable descending sort
      ra += ((vp < v) || tie) ? 1 : 0;   // position in stable ascending sort
    }
    if (rd < K_ACT) idx_act[b * K_ACT + rd] = t;
    if (ra < K_BKG) idx_bkg[b * K_BKG + ra] = t;
  }
}

// ---------------- per-(b,c) mean of top-83 cas values over time ----------------
__global__ void __launch_bounds__(256)
score_act_k(const float* __restrict__ cas, float* __restrict__ sa_pre) {
  __shared__ float col[T];
  __shared__ float partial[256];
  int b = blockIdx.x / C, c = blockIdx.x % C;
  for (int t = threadIdx.x; t < T; t += 256) col[t] = cas[((long)b * T + t) * C + c];
  __syncthreads();
  float s = 0.f;
  for (int t = threadIdx.x; t < T; t += 256) {
    float v = col[t];
    int rd = 0;
    for (int tp = 0; tp < T; ++tp) {
      float vp = col[tp];
      rd += ((vp > v) || ((vp == v) && (tp < t))) ? 1 : 0;
    }
    if (rd < K_ACT) s += v;
  }
  partial[threadIdx.x] = s;
  __syncthreads();
  for (int off = 128; off > 0; off >>= 1) {
    if ((int)threadIdx.x < off) partial[threadIdx.x] += partial[threadIdx.x + off];
    __syncthreads();
  }
  if (threadIdx.x == 0) sa_pre[b * C + c] = partial[0] / (float)K_ACT;
}

__global__ void score_bkg_k(const float* __restrict__ cas, const int* __restrict__ idx_bkg,
                            float* __restrict__ sb_pre) {
  int i = blockIdx.x * blockDim.x + threadIdx.x;
  if (i >= B * C) return;
  int b = i / C, c = i % C;
  float s = 0.f;
  for (int j = 0; j < K_BKG; ++j) {
    int t = idx_bkg[b * K_BKG + j];
    s += cas[((long)b * T + t) * C + c];
  }
  sb_pre[i] = s / (float)K_BKG;
}

__global__ void softmax_scores(const float* __restrict__ sa, const float* __restrict__ sb,
                               float* __restrict__ out_sa, float* __restrict__ out_sb) {
  int i = threadIdx.x;
  if (i >= 2 * B) return;
  const float* src = (i < B) ? (sa + i * C) : (sb + (i - B) * C);
  float* dst = (i < B) ? (out_sa + i * C) : (out_sb + (i - B) * C);
  float mx = src[0];
  for (int c = 1; c < C; ++c) mx = fmaxf(mx, src[c]);
  float s = 0.f;
  for (int c = 0; c < C; ++c) s += __expf(src[c] - mx);
  float inv = 1.f / s;
  for (int c = 0; c < C; ++c) dst[c] = __expf(src[c] - mx) * inv;
}

// ---------------- feature row gathers ----------------
__global__ void __launch_bounds__(256)
gather_feat(const float* __restrict__ feat, const int* __restrict__ idx_act,
            const int* __restrict__ idx_bkg, float* __restrict__ fa, float* __restrict__ fb) {
  int row = blockIdx.x;
  const float* src; float* dst;
  if (row < B * K_ACT) {
    int b = row / K_ACT;
    src = feat + ((long)b * T + idx_act[row]) * D;
    dst = fa + (long)row * D;
  } else {
    int r2 = row - B * K_ACT;
    int b = r2 / K_BKG;
    src = feat + ((long)b * T + idx_bkg[r2]) * D;
    dst = fb + (long)r2 * D;
  }
  const float4* s4 = (const float4*)src;
  float4* d4 = (float4*)dst;
  for (int k = threadIdx.x; k < D / 4; k += 256) d4[k] = s4[k];
}

// ---------------- launcher ----------------
extern "C" void kernel_launch(void* const* d_in, const int* in_sizes, int n_in,
                              void* d_out, int out_size, void* d_ws, size_t ws_size,
                              hipStream_t stream) {
  const float* x      = (const float*)d_in[0];
  const float* conv_w = (const float*)d_in[1];
  const float* conv_b = (const float*)d_in[2];
  const float* gamma  = (const float*)d_in[3];
  const float* beta   = (const float*)d_in[4];
  const float* mean   = (const float*)d_in[5];
  const float* var    = (const float*)d_in[6];
  const float* clsw   = (const float*)d_in[7];
  // d_in[8]=r_act, d_in[9]=r_bkg (compile-time constants here)

  float* out      = (float*)d_out;
  float* out_sa   = out;                                   // [16,20]
  float* out_sb   = out_sa + (long)B * C;                  // [16,20]
  float* out_fa   = out_sb + (long)B * C;                  // [16,83,2048]
  float* out_fb   = out_fa + (long)B * K_ACT * D;          // [16,187,2048]
  float* out_feat = out_fb + (long)B * K_BKG * D;          // [16,750,2048]
  float* out_csm  = out_feat + (long)B * T * D;            // [16,750,20]

  unsigned short* xp  = (unsigned short*)d_ws;             // XP_ROWS * D bf16
  unsigned short* wpt = xp + XP_ROWS * (long)D;            // 2048 * 6144 bf16
  float* scaleb = (float*)(wpt + (long)D * KDIM);
  float* offsb  = scaleb + D;
  float* cas    = offsb + D;                               // [B*T, C]
  float* fmag   = cas + (long)B * T * C;                   // [B*T]
  int* idx_act  = (int*)(fmag + (long)B * T);              // [B, 83]
  int* idx_bkg  = idx_act + B * K_ACT;                     // [B, 187]
  float* sa_pre = (float*)(idx_bkg + B * K_BKG);
  float* sb_pre = sa_pre + B * C;

  long xp_total = XP_ROWS * (long)D;
  prep_x<<<(unsigned)((xp_total + 255) / 256), 256, 0, stream>>>(x, xp);
  prep_w<<<(D * KDIM + 255) / 256, 256, 0, stream>>>(conv_w, wpt);
  prep_bn<<<(D + 255) / 256, 256, 0, stream>>>(gamma, beta, mean, var, conv_b, scaleb, offsb);

  dim3 g(B * 6, 8);   // 6 M-blocks of 128 rows per batch; 8 N-blocks of 256 cols
  conv_gemm<<<g, 512, 0, stream>>>(xp, wpt, scaleb, offsb, out_feat);

  cas_mag<<<(B * T + 7) / 8, 256, 0, stream>>>(out_feat, clsw, cas, fmag, out_csm);
  topk_idx<<<B, 256, 0, stream>>>(fmag, idx_act, idx_bkg);
  score_act_k<<<B * C, 256, 0, stream>>>(cas, sa_pre);
  score_bkg_k<<<1, 320, 0, stream>>>(cas, idx_bkg, sb_pre);
  softmax_scores<<<1, 32, 0, stream>>>(sa_pre, sb_pre, out_sa, out_sb);
  gather_feat<<<B * (K_ACT + K_BKG), 256, 0, stream>>>(out_feat, idx_act, idx_bkg, out_fa, out_fb);
}
